// OT_Loss_10084583211622
// MI455X (gfx1250) — matrix operand order repaired and checked by reading, not compiled, not run on hardware
//
#include <hip/hip_runtime.h>

typedef __attribute__((ext_vector_type(16))) _Float16 v16h;
typedef __attribute__((ext_vector_type(8)))  float    v8f;
typedef __attribute__((ext_vector_type(4)))  unsigned uv4;

#define SDIM   64
#define SS     4096
#define NPTS   512
#define NITER  100
#define MEPS   1e-16f

// ---- dynamic LDS layout (bytes) ----
#define OFF_KYT 0        // _Float16 Kyt[64][512]  (Ky transposed: [gy][i])
#define OFF_KXT 65536    // _Float16 Kxt[64][512]  (Kx transposed: [gx][i])
#define OFF_KX  131072   // _Float16 Kx [512][64]  (row-major [i][gx])
#define OFF_WT  196608   // _Float16 Wt [64][512]  (u[i]*Ky[i,gy], [gy][i])
#define OFF_V   262144   // float    v  [4096]
#define OFF_B   278528   // float    b  [4096]     (normed density)
#define OFF_VH  294912   // _Float16 Vh [4096]     (v as f16, [gy][gx])
#define OFF_U   303104   // float    u  [512]
#define OFF_KV  305152   // float    KV [512]      (K@v)
#define OFF_RED 307200   // float    red[32]
#define SMEM_BYTES 307328

union F16Frag { v16h h; uv4 q[2]; };

// A fragment (16x32 f16), ISA 7.12.2: lane holds row m (passed in, includes
// lane%16); VGPR j packs (k,k+1), k = 2j + 8*(lane/16) + (j>=4 ? 8 : 0).
// => halves [k0+8h, +8) and [k0+8h+16, +8): two 16B-aligned b128 LDS loads.
__device__ __forceinline__ v16h ldA(const _Float16* p, int stride, int row, int k0, int lane) {
  F16Frag f;
  const _Float16* r = p + row * stride + k0 + 8 * (lane >> 4);
  f.q[0] = *(const uv4*)(r);
  f.q[1] = *(const uv4*)(r + 16);
  return f.h;
}

// B fragment (32x16 f16): lane holds col n (passed in); VGPR j packs (k,k+1),
// k = 2j + 8*(lane/16) + (j>=4 ? 8 : 0).  Source stored transposed [n][k] =>
// halves [k0+8h, +8) and [k0+8h+16, +8): two b128 LDS loads.
__device__ __forceinline__ v16h ldB(const _Float16* p, int stride, int row, int k0, int lane) {
  F16Frag f;
  const _Float16* r = p + row * stride + k0 + 8 * (lane >> 4);
  f.q[0] = *(const uv4*)(r);
  f.q[1] = *(const uv4*)(r + 16);
  return f.h;
}

__device__ __forceinline__ float block_allreduce(float v, float* red, int tid) {
#pragma unroll
  for (int off = 16; off >= 1; off >>= 1) v += __shfl_xor(v, off, 32);
  if ((tid & 31) == 0) red[tid >> 5] = v;
  __syncthreads();
  if (tid == 0) {
    float s = 0.f;
#pragma unroll
    for (int w = 0; w < 16; ++w) s += red[w];
    red[16] = s;
  }
  __syncthreads();
  float r = red[16];
  __syncthreads();
  return r;
}

__global__ void __launch_bounds__(512)
sinkhorn_ot_kernel(const float* __restrict__ nd_all,
                   const float* __restrict__ ud_all,
                   const float* __restrict__ pts_all,
                   float* __restrict__ out) {
  extern __shared__ char smem[];
  _Float16* Kyt = (_Float16*)(smem + OFF_KYT);
  _Float16* Kxt = (_Float16*)(smem + OFF_KXT);
  _Float16* Kx  = (_Float16*)(smem + OFF_KX);
  _Float16* Wt  = (_Float16*)(smem + OFF_WT);
  float*    vL  = (float*)(smem + OFF_V);
  float*    bL  = (float*)(smem + OFF_B);
  _Float16* Vh  = (_Float16*)(smem + OFF_VH);
  float*    uL  = (float*)(smem + OFF_U);
  float*    KV  = (float*)(smem + OFF_KV);
  float*    red = (float*)(smem + OFF_RED);

  const int tid  = threadIdx.x;
  const int wave = tid >> 5;
  const int lane = tid & 31;
  const int l16  = lane & 15;
  const int half = lane >> 4;
  const int img  = blockIdx.x;

  const float* nd  = nd_all  + (size_t)img * SS;
  const float* ud  = ud_all  + (size_t)img * SS;
  const float* pts = pts_all + (size_t)img * NPTS * 2;

  // ---------------- setup: Ky/Kx (f16) from points, b, u0 ----------------
  // normalized coords: c = (g*8+4)/512*2-1 ; p' = p/512*2-1 ; dis = (p'-c)^2
  {
    const int i = tid;  // 512 threads, one point row each
    const float x = pts[2 * i]     * (2.0f / 512.0f) - 1.0f;
    const float y = pts[2 * i + 1] * (2.0f / 512.0f) - 1.0f;
#pragma unroll 4
    for (int g = 0; g < SDIM; ++g) {
      const float cg = (float)(g * 8 + 4) * (2.0f / 512.0f) - 1.0f;
      float dx = x - cg; dx *= dx;
      float dy = y - cg; dy *= dy;
      const _Float16 kx = (_Float16)__expf(-dx * 0.1f);  // REG = 10
      const _Float16 ky = (_Float16)__expf(-dy * 0.1f);
      Kx [i * SDIM + g] = kx;
      Kxt[g * NPTS + i] = kx;
      Kyt[g * NPTS + i] = ky;
    }
    uL[i] = 1.0f / (float)NPTS;
  }
  for (int j = tid; j < SS; j += 512) bL[j] = nd[j];
  __syncthreads();

  // ---------------- 100 Sinkhorn iterations, fully LDS-resident ----------
  for (int iter = 0; iter < NITER; ++iter) {
    // Wt[gy][i] = f16(u[i] * Ky[i,gy])
    for (int idx = tid; idx < SDIM * NPTS; idx += 512) {
      const int i = idx & (NPTS - 1);
      Wt[idx] = (_Float16)(uL[i] * (float)Kyt[idx]);
    }
    __syncthreads();

    // GEMM1: Z[gy,gx] = sum_i Wt[gy][i] * Kx[i][gx]   (A=Wt, B=Kxt, K=512)
    // then  v = b / (Z + eps)
    {
      const int m0 = (wave >> 2) * 16;   // gy tile
      const int n0 = (wave & 3) * 16;    // gx tile
      v8f c = {};
#pragma unroll
      for (int kt = 0; kt < NPTS / 32; ++kt) {
        v16h a = ldA(Wt,  NPTS, m0 + l16, kt * 32, lane);
        v16h b = ldB(Kxt, NPTS, n0 + l16, kt * 32, lane);
        c = __builtin_amdgcn_wmma_f32_16x16x32_f16(
                false, a, false, b, (short)0, c, false, false);
      }
      const int n = n0 + l16;
#pragma unroll
      for (int r = 0; r < 8; ++r) {
        const int m   = m0 + half * 8 + r;
        const int idx = m * SDIM + n;
        const float val = bL[idx] / (c[r] + MEPS);
        vL[idx] = val;
        Vh[idx] = (_Float16)val;
      }
    }
    __syncthreads();

    // GEMM2: T[i,gy] = sum_gx Kx[i][gx] * Vh[gy][gx]  (A=Kx, B=Vh, K=64)
    // then  KV[i] = sum_gy Ky[i,gy] * T[i,gy]
#pragma unroll
    for (int it = 0; it < 2; ++it) {
      const int i0 = (wave * 2 + it) * 16;
      // A fragments (Kx rows) are invariant across gy tiles: load once.
      const v16h a0 = ldA(Kx, SDIM, i0 + l16, 0,  lane);
      const v16h a1 = ldA(Kx, SDIM, i0 + l16, 32, lane);
      float rsum[8] = {0.f, 0.f, 0.f, 0.f, 0.f, 0.f, 0.f, 0.f};
#pragma unroll
      for (int nt = 0; nt < 4; ++nt) {
        const int n0 = nt * 16;
        v8f c = {};
        v16h b0 = ldB(Vh, SDIM, n0 + l16, 0,  lane);
        c = __builtin_amdgcn_wmma_f32_16x16x32_f16(
                false, a0, false, b0, (short)0, c, false, false);
        v16h b1 = ldB(Vh, SDIM, n0 + l16, 32, lane);
        c = __builtin_amdgcn_wmma_f32_16x16x32_f16(
                false, a1, false, b1, (short)0, c, false, false);
        const int n = n0 + l16;
#pragma unroll
        for (int r = 0; r < 8; ++r) {
          const int m = i0 + half * 8 + r;
          rsum[r] += c[r] * (float)Kyt[n * NPTS + m];
        }
      }
#pragma unroll
      for (int r = 0; r < 8; ++r) {
        float s = rsum[r];
#pragma unroll
        for (int off = 8; off >= 1; off >>= 1) s += __shfl_xor(s, off, 16);
        if (l16 == 0) KV[i0 + half * 8 + r] = s;
      }
    }
    __syncthreads();

    // u = a / (K@v + eps),  a = 1/N
    uL[tid] = (1.0f / (float)NPTS) / (KV[tid] + MEPS);
    __syncthreads();
  }

  // ---------------- epilogue ----------------
  // beta = REG*log(v+eps); ot = sum(b*beta); sc = sum(ud); t = sum(ud*beta)
  float sc_p = 0.f, t_p = 0.f, ot_p = 0.f;
  for (int j = tid; j < SS; j += 512) {
    const float beta = 10.0f * __logf(vL[j] + MEPS);
    const float sd = ud[j];
    sc_p += sd;
    t_p  += sd * beta;
    ot_p += bL[j] * beta;
  }
  const float sc = block_allreduce(sc_p, red, tid);
  const float t  = block_allreduce(t_p,  red, tid);
  const float ot = block_allreduce(ot_p, red, tid);

  const float denom = sc * sc + 1e-8f;
  const float im2   = t / denom;
  const float s1    = sc / denom;
  float loss_p = 0.f;
  for (int j = tid; j < SS; j += 512) {
    const float beta = 10.0f * __logf(vL[j] + MEPS);
    loss_p += ud[j] * (s1 * beta - im2);
  }
  const float loss = block_allreduce(loss_p, red, tid);

  // wd = sum_{i,gy,gx} (dy+dx) * u_i * Ky * Kx * v   (recompute dy/dx)
  float wd_p = 0.f;
  {
    const int i = tid;
    const float ui = uL[i];
    const float x = pts[2 * i]     * (2.0f / 512.0f) - 1.0f;
    const float y = pts[2 * i + 1] * (2.0f / 512.0f) - 1.0f;
    const _Float16* kxrow = Kx + i * SDIM;
    for (int gy = 0; gy < SDIM; ++gy) {
      const float cgy = (float)(gy * 8 + 4) * (2.0f / 512.0f) - 1.0f;
      float dy = y - cgy; dy *= dy;
      const float uk = ui * (float)Kyt[gy * NPTS + i];
      const float* vrow = vL + gy * SDIM;
#pragma unroll 4
      for (int gx = 0; gx < SDIM; ++gx) {
        const float cgx = (float)(gx * 8 + 4) * (2.0f / 512.0f) - 1.0f;
        float dx = x - cgx; dx *= dx;
        wd_p += (dy + dx) * uk * (float)kxrow[gx] * vrow[gx];
      }
    }
  }
  const float wd = block_allreduce(wd_p, red, tid);

  if (tid == 0) {
    atomicAdd(out + 0, loss);
    atomicAdd(out + 1, wd);
    atomicAdd(out + 2, ot);
  }
}

__global__ void zero_out_kernel(float* out, int n) {
  const int i = threadIdx.x;
  if (i < n) out[i] = 0.f;
}

extern "C" void kernel_launch(void* const* d_in, const int* in_sizes, int n_in,
                              void* d_out, int out_size, void* d_ws, size_t ws_size,
                              hipStream_t stream) {
  const float* nd  = (const float*)d_in[0];  // normed_density  (B,1,64,64)
  const float* ud  = (const float*)d_in[1];  // unnormed_density
  const float* pts = (const float*)d_in[2];  // points (B,512,2)
  float* out = (float*)d_out;                // (loss, wd, ot_obj)

  const int B = in_sizes[0] / SS;

  zero_out_kernel<<<1, 32, 0, stream>>>(out, out_size);
  sinkhorn_ot_kernel<<<B, 512, SMEM_BYTES, stream>>>(nd, ud, pts, out);
}
